// SimilarityEntropy_70635032150357
// MI455X (gfx1250) — matrix operand order repaired and checked by reading, not compiled, not run on hardware
//
#include <hip/hip_runtime.h>
#include <hip/hip_bf16.h>
#include <cstdint>
#include <cstddef>

#define N_PTS 8192
#define DIM   768
#define DIAG_PEN 100.0f
#define SLICE_K 128
#define WAVES 8
#define COLS_GRP 64
#define TILES 4

typedef __bf16 bf16_t;
typedef bf16_t v16bf __attribute__((ext_vector_type(16)));
typedef float  v8f   __attribute__((ext_vector_type(8)));

union FragAB { v16bf v; uint4 q[2]; };

__device__ __forceinline__ unsigned short f2bf(float f) {
  unsigned u = __float_as_uint(f);
  unsigned r = u + 0x7FFFu + ((u >> 16) & 1u);   // round-to-nearest-even
  return (unsigned short)(r >> 16);
}
__device__ __forceinline__ float bf2f(unsigned short b) {
  return __uint_as_float(((unsigned)b) << 16);
}

// ---------------- prep: bf16 hi/lo split + row squared norms ----------------
__global__ __launch_bounds__(256) void prep_kernel(
    const float* __restrict__ x,
    unsigned short* __restrict__ xhi,
    unsigned short* __restrict__ xlo,
    float* __restrict__ sq) {
  __shared__ float red[256];
  const int row = blockIdx.x;
  const float* xr = x + (size_t)row * DIM;
  float acc = 0.f;
  for (int e = threadIdx.x; e < DIM; e += 256) {
    float f = xr[e];
    unsigned short h = f2bf(f);
    unsigned short l = f2bf(f - bf2f(h));
    xhi[(size_t)row * DIM + e] = h;
    xlo[(size_t)row * DIM + e] = l;
    acc += f * f;
  }
  red[threadIdx.x] = acc;
  __syncthreads();
  for (int s = 128; s > 0; s >>= 1) {
    if (threadIdx.x < s) red[threadIdx.x] += red[threadIdx.x + s];
    __syncthreads();
  }
  if (threadIdx.x == 0) sq[row] = red[0];
}

// ------------- fused gram (split-bf16 WMMA) + streaming entropy -------------
__global__ __launch_bounds__(256) void sim_entropy_kernel(
    const unsigned short* __restrict__ xhi,
    const unsigned short* __restrict__ xlo,
    const float* __restrict__ sq,
    const float* __restrict__ tau_p,
    float* __restrict__ partials) {
  __shared__ __align__(32) unsigned short ldsHi[COLS_GRP * SLICE_K];
  __shared__ __align__(32) unsigned short ldsLo[COLS_GRP * SLICE_K];
  __shared__ float waveSums[WAVES];

  const float tau = tau_p[0];
  const int tid    = threadIdx.x;
  const int wave   = tid >> 5;
  const int lane   = tid & 31;
  const int laneLo = lane & 15;
  const int laneHi = lane >> 4;                 // 0 or 1
  const int rowBase = (blockIdx.x * WAVES + wave) * 16;
  const int myRow   = rowBase + laneLo;         // A-fragment source row

  // sq for accumulator slot v: row = rowBase + v + 8*laneHi  (C-matrix layout)
  float sqi[8];
  #pragma unroll
  for (int v = 0; v < 8; ++v) sqi[v] = sq[rowBase + v + 8 * laneHi];

  float mrow[8], zrow[8], srow[8];
  #pragma unroll
  for (int v = 0; v < 8; ++v) { mrow[v] = -1e30f; zrow[v] = 0.f; srow[v] = 0.f; }

  const unsigned short* aHiRow = xhi + (size_t)myRow * DIM;
  const unsigned short* aLoRow = xlo + (size_t)myRow * DIM;

  for (int colBase = 0; colBase < N_PTS; colBase += COLS_GRP) {
    v8f acc[TILES];
    #pragma unroll
    for (int t = 0; t < TILES; ++t)
      #pragma unroll
      for (int e = 0; e < 8; ++e) acc[t][e] = 0.f;

    for (int kOuter = 0; kOuter < DIM; kOuter += SLICE_K) {
      __syncthreads();
      // stage 64 cols x 128 K of hi/lo (col-major rows) into LDS: 4 uint4/thread/array
      #pragma unroll
      for (int i = 0; i < 4; ++i) {
        int idx = tid * 4 + i;                 // 0..1023
        int c   = idx >> 4;                    // 0..63
        int q8  = idx & 15;                    // 8-ushort chunk
        size_t gOff = (size_t)(colBase + c) * DIM + kOuter + q8 * 8;
        int lOff = c * SLICE_K + q8 * 8;
        *(uint4*)(&ldsHi[lOff]) = *(const uint4*)(&xhi[gOff]);
        *(uint4*)(&ldsLo[lOff]) = *(const uint4*)(&xlo[gOff]);
      }
      __syncthreads();

      #pragma unroll
      for (int kc = 0; kc < SLICE_K / 32; ++kc) {
        const int kBase = kOuter + kc * 32;
        // A fragment (ISA 16-bit A 16x32 layout): lane holds row l%16,
        // halfs[0..7] = K[kBase + laneHi*8 ..], halfs[8..15] = K[kBase+16+laneHi*8 ..]
        FragAB aHi, aLo;
        const int kA = kBase + laneHi * 8;
        aHi.q[0] = *(const uint4*)(aHiRow + kA);
        aHi.q[1] = *(const uint4*)(aHiRow + kA + 16);
        aLo.q[0] = *(const uint4*)(aLoRow + kA);
        aLo.q[1] = *(const uint4*)(aLoRow + kA + 16);

        #pragma unroll
        for (int t = 0; t < TILES; ++t) {
          // B fragment: lane holds column l%16, halfs[0..7] = K[kc*32 + laneHi*8 ..], halfs[8..15] = K[kc*32+16+laneHi*8 ..]
          const int lOff = (t * 16 + laneLo) * SLICE_K + kc * 32 + laneHi * 8;
          FragAB bHi, bLo;
          bHi.q[0] = *(const uint4*)(&ldsHi[lOff]);
          bHi.q[1] = *(const uint4*)(&ldsHi[lOff + 16]);
          bLo.q[0] = *(const uint4*)(&ldsLo[lOff]);
          bLo.q[1] = *(const uint4*)(&ldsLo[lOff + 16]);
          // split-precision gram: hi*hi + hi*lo + lo*hi, f32 accumulate
          acc[t] = __builtin_amdgcn_wmma_f32_16x16x32_bf16(false, aHi.v, false, bHi.v, (short)0, acc[t], false, false);
          acc[t] = __builtin_amdgcn_wmma_f32_16x16x32_bf16(false, aHi.v, false, bLo.v, (short)0, acc[t], false, false);
          acc[t] = __builtin_amdgcn_wmma_f32_16x16x32_bf16(false, aLo.v, false, bHi.v, (short)0, acc[t], false, false);
        }
      }
    }

    // streaming softmax-entropy update (per lane: cols j with j%16 == laneLo)
    #pragma unroll
    for (int t = 0; t < TILES; ++t) {
      const int jCol = colBase + t * 16 + laneLo;
      const float sqj = sq[jCol];
      #pragma unroll
      for (int v = 0; v < 8; ++v) {
        const int iRow = rowBase + v + 8 * laneHi;
        float simv = 2.f * acc[t][v] - sqi[v] - sqj;
        if (iRow == jCol) simv -= DIAG_PEN;
        const float s  = tau * simv;
        const float m2 = fmaxf(mrow[v], s);
        const float dm = mrow[v] - m2;
        const float ds = s - m2;
        const float e1 = __expf(dm);
        const float e2 = __expf(ds);
        srow[v] = (srow[v] + dm * zrow[v]) * e1 + ds * e2;
        zrow[v] = zrow[v] * e1 + e2;
        mrow[v] = m2;
      }
    }
  }

  // merge partial stats across the 16 lanes holding each row (logsumexp merge)
  #pragma unroll
  for (int v = 0; v < 8; ++v) {
    #pragma unroll
    for (int mask = 1; mask <= 8; mask <<= 1) {
      float mo = __shfl_xor(mrow[v], mask, 32);
      float zo = __shfl_xor(zrow[v], mask, 32);
      float so = __shfl_xor(srow[v], mask, 32);
      float m2 = fmaxf(mrow[v], mo);
      float e1 = __expf(mrow[v] - m2);
      float e2 = __expf(mo - m2);
      srow[v] = (srow[v] + (mrow[v] - m2) * zrow[v]) * e1 +
                (so      + (mo      - m2) * zo)      * e2;
      zrow[v] = zrow[v] * e1 + zo * e2;
      mrow[v] = m2;
    }
  }

  // entropy_i = log Z - S1/Z ; sum this wave's 16 rows
  float entSum = 0.f;
  #pragma unroll
  for (int v = 0; v < 8; ++v) entSum += logf(zrow[v]) - srow[v] / zrow[v];
  entSum += __shfl_xor(entSum, 16, 32);         // combine the two 8-row halves

  if (lane == 0) waveSums[wave] = entSum;
  __syncthreads();
  if (tid == 0) {
    float tot = 0.f;
    for (int w = 0; w < WAVES; ++w) tot += waveSums[w];
    partials[blockIdx.x] = tot;
  }
}

// -------- deterministic finalize (fixed summation order, no atomics) --------
__global__ void finalize_kernel(const float* __restrict__ partials,
                                const float* __restrict__ coef_p,
                                float* __restrict__ out, int nPart) {
  if (threadIdx.x == 0 && blockIdx.x == 0) {
    float tot = 0.f;
    for (int i = 0; i < nPart; ++i) tot += partials[i];
    out[0] = coef_p[0] * tot / (float)N_PTS;
  }
}

extern "C" void kernel_launch(void* const* d_in, const int* in_sizes, int n_in,
                              void* d_out, int out_size, void* d_ws, size_t ws_size,
                              hipStream_t stream) {
  (void)in_sizes; (void)n_in; (void)out_size; (void)ws_size;
  const float* x    = (const float*)d_in[0];
  const float* coef = (const float*)d_in[1];
  const float* tau  = (const float*)d_in[2];
  float* out = (float*)d_out;

  unsigned char* ws = (unsigned char*)d_ws;
  unsigned short* xhi = (unsigned short*)ws;                         // 12.6 MB
  unsigned short* xlo = xhi + (size_t)N_PTS * DIM;                   // 12.6 MB
  float* sq       = (float*)(xlo + (size_t)N_PTS * DIM);             // 32 KB
  float* partials = sq + N_PTS;                                      // 64 floats

  prep_kernel<<<N_PTS, 256, 0, stream>>>(x, xhi, xlo, sq);
  sim_entropy_kernel<<<N_PTS / (WAVES * 16), 256, 0, stream>>>(xhi, xlo, sq, tau, partials);
  finalize_kernel<<<1, 64, 0, stream>>>(partials, coef, out, N_PTS / (WAVES * 16));
}
